// RoIHeads_16982300688572
// MI455X (gfx1250) — matrix operand order repaired and checked by reading, not compiled, not run on hardware
//
#include <hip/hip_runtime.h>
#include <hip/hip_bf16.h>
#include <math.h>

// ---------------- problem constants ----------------
#define NUM_IMAGES 2
#define N_PROP 1000
#define NUM_CLASSES 91
#define IN_CH 12544
#define REP 1024
#define SCORE_TH 0.05f
#define NMS_TH 0.5f
#define DET_PER_IMG 100
#define IMG_W 800.0f
#define IMG_H 800.0f
#define BBOX_CLIP 4.135166556742356f   // log(1000/16)
#define NCLS_FG (NUM_CLASSES - 1)      // 90
#define NPROPS_ALL (NUM_IMAGES * N_PROP) // 2000
#define FLAT_PER_IMG (NCLS_FG * DET_PER_IMG) // 9000

typedef __attribute__((ext_vector_type(16))) _Float16 v16h;
typedef __attribute__((ext_vector_type(8)))  _Float16 v8h;
typedef __attribute__((ext_vector_type(8)))  float    v8f;

#define NEG_INF (-__builtin_inff())

// ==================================================================
// Tiled WMMA GEMM:  C[M,N] = act(A[M,K] @ B[K,N] + bias[N])
// block = 256 threads = 8 waves; block tile 64(M) x 128(N); K step 32.
// Waves arranged 2(M) x 4(N); each wave owns a 32x32 output tile =
// 4 accumulators -> 4 v_wmma_f32_16x16x32_f16 per K step fed by
// 8 ds_load_b128 (2 A frags + 2 B frags, 2 loads each).
// A staged f16 [64][32] (stride 40 halves = 80B, 16B-aligned chunks),
// B staged transposed f16 [128 n][32 k] so fragments are contiguous.
// FULL_N=true removes all N guards (compile-time) for the hot GEMMs.
// ==================================================================
#define LDT 40  // padded LDS row stride in halves

template <typename AT, bool RELU, typename OT, bool FULL_N>
__global__ __launch_bounds__(256) void gemm_wmma_kernel(
    const AT* __restrict__ A, const float* __restrict__ B,
    const float* __restrict__ bias, OT* __restrict__ C,
    int M, int N, int K) {
  __shared__ _Float16 As[64 * LDT];
  __shared__ _Float16 Bs[128 * LDT];

  const int tid  = threadIdx.x;
  const int lane = tid & 31;
  const int wid  = tid >> 5;
  const int blockM = blockIdx.y * 64;
  const int blockN = blockIdx.x * 128;

  const int m0 = (wid & 1) * 32;   // wave M offset in block tile
  const int n0 = (wid >> 1) * 32;  // wave N offset in block tile

  v8f acc00 = {}, acc01 = {}, acc10 = {}, acc11 = {};

  // A fragment addressing (16-bit A 16x32 layout):
  // lane<16: K 0..7 in v0..3, K 16..23 in v4..7 ; lane>=16: +8 on K base
  const int rowA0 = m0 + (lane & 15);
  const int rowA1 = rowA0 + 16;
  const int kA0   = (lane >> 4) * 8;           // 0 or 8 (halves)
  // B fragment addressing (16-bit B 32x16): same K order as A (lane>=16: +8 on K base)
  const int colB  = lane & 15;
  const int kB0   = (lane >> 4) * 8;           // 0 or 8 (halves)

  for (int kb = 0; kb < K; kb += 32) {
    // ---- stage A tile: 64 rows x 32 k (4 threads/row, 8 halves each) ----
    {
      const int row  = tid >> 2;
      const int kc   = (tid & 3) * 8;
      const int grow = blockM + row;
      v8h st;
      if (grow < M) {
        if constexpr (sizeof(AT) == 4) {
          const float4* p = (const float4*)(A + (size_t)grow * K + kb + kc);
          float4 v0 = p[0], v1 = p[1];
          st[0] = (_Float16)v0.x; st[1] = (_Float16)v0.y;
          st[2] = (_Float16)v0.z; st[3] = (_Float16)v0.w;
          st[4] = (_Float16)v1.x; st[5] = (_Float16)v1.y;
          st[6] = (_Float16)v1.z; st[7] = (_Float16)v1.w;
        } else {
          st = *(const v8h*)(A + (size_t)grow * K + kb + kc);
        }
        if (kb + 32 < K)
          __builtin_prefetch(A + (size_t)grow * K + kb + 32 + kc, 0, 3);
      } else {
#pragma unroll
        for (int j = 0; j < 8; ++j) st[j] = (_Float16)0.0f;
      }
      *(v8h*)&As[row * LDT + kc] = st;
    }
    // ---- stage B tile transposed: Bs[n][k], 32 k x 128 n ----
    {
      const int k  = tid >> 3;          // 0..31
      const int nc = (tid & 7) * 16;    // 0..112
      if constexpr (FULL_N) {
        const float4* p = (const float4*)(B + (size_t)(kb + k) * N + blockN + nc);
        float4 q0 = p[0], q1 = p[1], q2 = p[2], q3 = p[3];
        _Float16 hv[16];
        hv[0]  = (_Float16)q0.x; hv[1]  = (_Float16)q0.y;
        hv[2]  = (_Float16)q0.z; hv[3]  = (_Float16)q0.w;
        hv[4]  = (_Float16)q1.x; hv[5]  = (_Float16)q1.y;
        hv[6]  = (_Float16)q1.z; hv[7]  = (_Float16)q1.w;
        hv[8]  = (_Float16)q2.x; hv[9]  = (_Float16)q2.y;
        hv[10] = (_Float16)q2.z; hv[11] = (_Float16)q2.w;
        hv[12] = (_Float16)q3.x; hv[13] = (_Float16)q3.y;
        hv[14] = (_Float16)q3.z; hv[15] = (_Float16)q3.w;
#pragma unroll
        for (int j = 0; j < 16; ++j) Bs[(nc + j) * LDT + k] = hv[j];
        if (kb + 32 < K)
          __builtin_prefetch(B + (size_t)(kb + 32 + k) * N + blockN + nc, 0, 3);
      } else {
#pragma unroll
        for (int j = 0; j < 16; ++j) {
          const int gcol = blockN + nc + j;
          float v = (gcol < N) ? B[(size_t)(kb + k) * N + gcol] : 0.0f;
          Bs[(nc + j) * LDT + k] = (_Float16)v;
        }
      }
    }
    __syncthreads();

    // ---- fragments + WMMA ----
    v16h af0, af1, bf0, bf1;
    {
      v8h lo = *(const v8h*)&As[rowA0 * LDT + kA0];
      v8h hi = *(const v8h*)&As[rowA0 * LDT + kA0 + 16];
#pragma unroll
      for (int j = 0; j < 8; ++j) { af0[j] = lo[j]; af0[j + 8] = hi[j]; }
    }
    {
      v8h lo = *(const v8h*)&As[rowA1 * LDT + kA0];
      v8h hi = *(const v8h*)&As[rowA1 * LDT + kA0 + 16];
#pragma unroll
      for (int j = 0; j < 8; ++j) { af1[j] = lo[j]; af1[j + 8] = hi[j]; }
    }
    {
      v8h lo = *(const v8h*)&Bs[(n0 + colB) * LDT + kB0];
      v8h hi = *(const v8h*)&Bs[(n0 + colB) * LDT + kB0 + 16];
#pragma unroll
      for (int j = 0; j < 8; ++j) { bf0[j] = lo[j]; bf0[j + 8] = hi[j]; }
    }
    {
      v8h lo = *(const v8h*)&Bs[(n0 + 16 + colB) * LDT + kB0];
      v8h hi = *(const v8h*)&Bs[(n0 + 16 + colB) * LDT + kB0 + 16];
#pragma unroll
      for (int j = 0; j < 8; ++j) { bf1[j] = lo[j]; bf1[j + 8] = hi[j]; }
    }

    acc00 = __builtin_amdgcn_wmma_f32_16x16x32_f16(false, af0, false, bf0,
                                                   (short)0, acc00, false, false);
    acc01 = __builtin_amdgcn_wmma_f32_16x16x32_f16(false, af0, false, bf1,
                                                   (short)0, acc01, false, false);
    acc10 = __builtin_amdgcn_wmma_f32_16x16x32_f16(false, af1, false, bf0,
                                                   (short)0, acc10, false, false);
    acc11 = __builtin_amdgcn_wmma_f32_16x16x32_f16(false, af1, false, bf1,
                                                   (short)0, acc11, false, false);
    __syncthreads();
  }

  // ---- epilogue: C/D layout -> VGPR r: lanes0-15 M=r, lanes16-31 M=8+r ----
  const int colL  = lane & 15;
  const int rbase = (lane >> 4) * 8;
#pragma unroll
  for (int mf = 0; mf < 2; ++mf) {
    const v8f* accs[2] = { mf ? &acc10 : &acc00, mf ? &acc11 : &acc01 };
#pragma unroll
    for (int r = 0; r < 8; ++r) {
      const int grow = blockM + m0 + mf * 16 + rbase + r;
      if (grow < M) {
#pragma unroll
        for (int nf = 0; nf < 2; ++nf) {
          const int gcol = blockN + n0 + nf * 16 + colL;
          if (FULL_N || gcol < N) {
            float v = (*accs[nf])[r] + bias[gcol];
            if (RELU) v = fmaxf(v, 0.0f);
            C[(size_t)grow * N + gcol] = (OT)v;
          }
        }
      }
    }
  }
}

// ==================================================================
// Softmax(91) + Faster-RCNN box decode. One block per proposal.
// Writes class-major scores_c [I,90,N] and boxes_c [I,90,N,4].
// ==================================================================
__global__ __launch_bounds__(128) void softmax_decode_kernel(
    const float* __restrict__ logits, const float* __restrict__ breg,
    const float* __restrict__ proposals,
    float* __restrict__ scores_c, float* __restrict__ boxes_c) {
  const int p   = blockIdx.x;              // 0..1999
  const int tid = threadIdx.x;
  __shared__ float sl[NUM_CLASSES];
  __shared__ float smax, ssum;

  if (tid < NUM_CLASSES) sl[tid] = logits[(size_t)p * NUM_CLASSES + tid];
  __syncthreads();
  if (tid == 0) {
    float m = sl[0];
    for (int i = 1; i < NUM_CLASSES; ++i) m = fmaxf(m, sl[i]);
    smax = m;
  }
  __syncthreads();
  if (tid < NUM_CLASSES) sl[tid] = expf(sl[tid] - smax);
  __syncthreads();
  if (tid == 0) {
    float s = 0.0f;
    for (int i = 0; i < NUM_CLASSES; ++i) s += sl[i];
    ssum = s;
  }
  __syncthreads();

  if (tid >= 1 && tid < NUM_CLASSES) {
    const int img = p / N_PROP;
    const int n   = p % N_PROP;
    const float x1 = proposals[p * 4 + 0], y1 = proposals[p * 4 + 1];
    const float x2 = proposals[p * 4 + 2], y2 = proposals[p * 4 + 3];
    const float pw = x2 - x1, ph = y2 - y1;
    const float px = x1 + 0.5f * pw, py = y1 + 0.5f * ph;

    const float* d = &breg[(size_t)p * (NUM_CLASSES * 4) + tid * 4];
    const float dx = d[0] * 0.1f;
    const float dy = d[1] * 0.1f;
    const float dw = fminf(d[2] * 0.2f, BBOX_CLIP);
    const float dh = fminf(d[3] * 0.2f, BBOX_CLIP);
    const float cx = dx * pw + px;
    const float cy = dy * ph + py;
    const float w  = expf(dw) * pw;
    const float h  = expf(dh) * ph;
    const float bx1 = fminf(fmaxf(cx - 0.5f * w, 0.0f), IMG_W);
    const float by1 = fminf(fmaxf(cy - 0.5f * h, 0.0f), IMG_H);
    const float bx2 = fminf(fmaxf(cx + 0.5f * w, 0.0f), IMG_W);
    const float by2 = fminf(fmaxf(cy + 0.5f * h, 0.0f), IMG_H);

    const size_t ci = (size_t)(img * NCLS_FG + (tid - 1)) * N_PROP + n;
    scores_c[ci] = sl[tid] / ssum;
    float* bo = &boxes_c[ci * 4];
    bo[0] = bx1; bo[1] = by1; bo[2] = bx2; bo[3] = by2;
  }
}

// ==================================================================
// Per-(image,class) greedy NMS. Block per (img,cls): bitonic sort of
// 1024 (score desc, index asc on ties — matches stable argsort(-s)),
// then sequential greedy suppression, then compact top-100.
// ==================================================================
#define NS 1024
__global__ __launch_bounds__(256) void nms_kernel(
    const float* __restrict__ scores_c, const float* __restrict__ boxes_c,
    float* __restrict__ cls_scores, float* __restrict__ cls_boxes) {
  const int blk = blockIdx.x;          // 0..179
  const int img = blk / NCLS_FG;
  const int c   = blk % NCLS_FG;
  const int tid = threadIdx.x;

  __shared__ float key[NS];
  __shared__ int   sidx[NS];
  __shared__ float bX1[NS], bY1[NS], bX2[NS], bY2[NS], bAr[NS];
  __shared__ unsigned char keep[NS];
  __shared__ int nv_s;

  const size_t base = (size_t)(img * NCLS_FG + c) * N_PROP;

  for (int e = tid; e < NS; e += 256) {
    if (e < N_PROP) {
      float s = scores_c[base + e];
      key[e]  = (s > SCORE_TH) ? s : NEG_INF;
      sidx[e] = e;
    } else {
      key[e]  = NEG_INF;
      sidx[e] = e;
    }
  }
  __syncthreads();

  // bitonic sort: descending by key, ascending by index on ties
  for (int k = 2; k <= NS; k <<= 1) {
    for (int j = k >> 1; j > 0; j >>= 1) {
      for (int i = tid; i < NS; i += 256) {
        const int ij = i ^ j;
        if (ij > i) {
          const float ka = key[i], kbv = key[ij];
          const int ia = sidx[i], ib = sidx[ij];
          const bool aFirst = (ka > kbv) || (ka == kbv && ia < ib);
          const bool descRegion = ((i & k) == 0);
          if (descRegion ? !aFirst : aFirst) {
            key[i] = kbv; key[ij] = ka;
            sidx[i] = ib; sidx[ij] = ia;
          }
        }
      }
      __syncthreads();
    }
  }

  // gather sorted boxes, areas, init keep
  for (int e = tid; e < NS; e += 256) {
    const int o = sidx[e];
    float x1 = 0.f, y1 = 0.f, x2 = 0.f, y2 = 0.f;
    if (o < N_PROP) {
      const float* b = &boxes_c[(base + o) * 4];
      x1 = b[0]; y1 = b[1]; x2 = b[2]; y2 = b[3];
    }
    bX1[e] = x1; bY1[e] = y1; bX2[e] = x2; bY2[e] = y2;
    bAr[e] = (x2 - x1) * (y2 - y1);
    keep[e] = (key[e] > -1e30f) ? 1 : 0;
  }
  if (tid == 0) {
    int nv = 0;
    while (nv < N_PROP && key[nv] > -1e30f) nv++;
    nv_s = nv;
  }
  __syncthreads();
  const int nv = nv_s;

  // greedy suppression (sequential over i, parallel over j)
  for (int i = 0; i < nv; ++i) {
    __syncthreads();
    if (keep[i]) {
      const float ix1 = bX1[i], iy1 = bY1[i], ix2 = bX2[i], iy2 = bY2[i];
      const float iar = bAr[i];
      for (int j = i + 1 + tid; j < nv; j += 256) {
        if (keep[j]) {
          const float xx1 = fmaxf(ix1, bX1[j]);
          const float yy1 = fmaxf(iy1, bY1[j]);
          const float xx2 = fminf(ix2, bX2[j]);
          const float yy2 = fminf(iy2, bY2[j]);
          const float w = fmaxf(xx2 - xx1, 0.0f);
          const float h = fmaxf(yy2 - yy1, 0.0f);
          const float inter = w * h;
          const float iou = inter / (iar + bAr[j] - inter + 1e-12f);
          if (iou > NMS_TH) keep[j] = 0;
        }
      }
    }
  }
  __syncthreads();

  // compact: first 100 kept (sorted order), pad with -inf
  if (tid == 0) {
    const size_t os = (size_t)(img * NCLS_FG + c) * DET_PER_IMG;
    int cnt = 0;
    for (int e = 0; e < nv && cnt < DET_PER_IMG; ++e) {
      if (keep[e]) {
        cls_scores[os + cnt] = key[e];
        float* bo = &cls_boxes[(os + cnt) * 4];
        bo[0] = bX1[e]; bo[1] = bY1[e]; bo[2] = bX2[e]; bo[3] = bY2[e];
        cnt++;
      }
    }
    for (; cnt < DET_PER_IMG; ++cnt) {
      cls_scores[os + cnt] = NEG_INF;
      float* bo = &cls_boxes[(os + cnt) * 4];
      bo[0] = 0.f; bo[1] = 0.f; bo[2] = 0.f; bo[3] = 0.f;
    }
  }
}

// ==================================================================
// Global top-100 per image over 9000 candidates (iterative argmax,
// index tie-break matches lax.top_k). Writes final output layout:
// boxes[2*100*4] | scores[2*100] | labels[2*100] (labels as float).
// ==================================================================
__global__ __launch_bounds__(256) void topk_kernel(
    const float* __restrict__ cls_scores, const float* __restrict__ cls_boxes,
    float* __restrict__ out) {
  const int img = blockIdx.x;
  const int tid = threadIdx.x;
  __shared__ float s[FLAT_PER_IMG];
  __shared__ float rv[256];
  __shared__ int   ri[256];

  for (int t = tid; t < FLAT_PER_IMG; t += 256)
    s[t] = cls_scores[(size_t)img * FLAT_PER_IMG + t];
  __syncthreads();

  for (int r = 0; r < DET_PER_IMG; ++r) {
    float bv = NEG_INF;
    int   bi = FLAT_PER_IMG;
    for (int t = tid; t < FLAT_PER_IMG; t += 256) {
      const float v = s[t];
      if (v > bv || (v == bv && t < bi)) { bv = v; bi = t; }
    }
    rv[tid] = bv; ri[tid] = bi;
    __syncthreads();
    for (int off = 128; off > 0; off >>= 1) {
      if (tid < off) {
        const float v2 = rv[tid + off];
        const int   i2 = ri[tid + off];
        if (v2 > rv[tid] || (v2 == rv[tid] && i2 < ri[tid])) {
          rv[tid] = v2; ri[tid] = i2;
        }
      }
      __syncthreads();
    }
    if (tid == 0) {
      const int   b  = ri[0];
      const float sv = rv[0];
      const bool  ok = (sv > -1e30f);
      float* ob = &out[(size_t)(img * DET_PER_IMG + r) * 4];
      if (ok) {
        const float* bb = &cls_boxes[((size_t)img * FLAT_PER_IMG + b) * 4];
        ob[0] = bb[0]; ob[1] = bb[1]; ob[2] = bb[2]; ob[3] = bb[3];
      } else {
        ob[0] = 0.f; ob[1] = 0.f; ob[2] = 0.f; ob[3] = 0.f;
      }
      out[NUM_IMAGES * DET_PER_IMG * 4 + img * DET_PER_IMG + r] = ok ? sv : 0.0f;
      out[NUM_IMAGES * DET_PER_IMG * 5 + img * DET_PER_IMG + r] =
          ok ? (float)(b / DET_PER_IMG + 1) : 0.0f;
      s[b] = NEG_INF;
    }
    __syncthreads();
  }
}

// ==================================================================
extern "C" void kernel_launch(void* const* d_in, const int* in_sizes, int n_in,
                              void* d_out, int out_size, void* d_ws, size_t ws_size,
                              hipStream_t stream) {
  const float* box_features = (const float*)d_in[0];
  const float* proposals    = (const float*)d_in[1];
  const float* W6 = (const float*)d_in[2];
  const float* b6 = (const float*)d_in[3];
  const float* W7 = (const float*)d_in[4];
  const float* b7 = (const float*)d_in[5];
  const float* Wc = (const float*)d_in[6];
  const float* bc = (const float*)d_in[7];
  const float* Wb = (const float*)d_in[8];
  const float* bb = (const float*)d_in[9];
  float* out = (float*)d_out;

  // workspace carve-up (all offsets 16B aligned)
  char* ws = (char*)d_ws;
  size_t off = 0;
  _Float16* x1 = (_Float16*)(ws + off); off += (size_t)NPROPS_ALL * REP * 2;      // 4 MB
  _Float16* x2 = (_Float16*)(ws + off); off += (size_t)NPROPS_ALL * REP * 2;      // 4 MB
  float* logits = (float*)(ws + off);   off += (size_t)NPROPS_ALL * NUM_CLASSES * 4;
  float* breg   = (float*)(ws + off);   off += (size_t)NPROPS_ALL * NUM_CLASSES * 4 * 4;
  float* scores_c = (float*)(ws + off); off += (size_t)NUM_IMAGES * NCLS_FG * N_PROP * 4;
  float* boxes_c  = (float*)(ws + off); off += (size_t)NUM_IMAGES * NCLS_FG * N_PROP * 4 * 4;
  float* cls_scores = (float*)(ws + off); off += (size_t)NUM_IMAGES * FLAT_PER_IMG * 4;
  float* cls_boxes  = (float*)(ws + off); off += (size_t)NUM_IMAGES * FLAT_PER_IMG * 4 * 4;
  (void)ws_size; (void)in_sizes; (void)n_in; (void)out_size;

  const dim3 blk(256);
  // GEMM1: relu(features @ W6 + b6) -> x1 (f16)   M=2000 K=12544 N=1024
  gemm_wmma_kernel<float, true, _Float16, true>
      <<<dim3(REP / 128, (NPROPS_ALL + 63) / 64), blk, 0, stream>>>(
          box_features, W6, b6, x1, NPROPS_ALL, REP, IN_CH);
  // GEMM2: relu(x1 @ W7 + b7) -> x2 (f16)         M=2000 K=1024 N=1024
  gemm_wmma_kernel<_Float16, true, _Float16, true>
      <<<dim3(REP / 128, (NPROPS_ALL + 63) / 64), blk, 0, stream>>>(
          x1, W7, b7, x2, NPROPS_ALL, REP, REP);
  // GEMM3: x2 @ Wc + bc -> logits (f32)           N=91
  gemm_wmma_kernel<_Float16, false, float, false>
      <<<dim3((NUM_CLASSES + 127) / 128, (NPROPS_ALL + 63) / 64), blk, 0, stream>>>(
          x2, Wc, bc, logits, NPROPS_ALL, NUM_CLASSES, REP);
  // GEMM4: x2 @ Wb + bb -> breg (f32)             N=364
  gemm_wmma_kernel<_Float16, false, float, false>
      <<<dim3((NUM_CLASSES * 4 + 127) / 128, (NPROPS_ALL + 63) / 64), blk, 0, stream>>>(
          x2, Wb, bb, breg, NPROPS_ALL, NUM_CLASSES * 4, REP);

  // softmax + decode
  softmax_decode_kernel<<<dim3(NPROPS_ALL), dim3(128), 0, stream>>>(
      logits, breg, proposals, scores_c, boxes_c);

  // per-class NMS (180 blocks)
  nms_kernel<<<dim3(NUM_IMAGES * NCLS_FG), blk, 0, stream>>>(
      scores_c, boxes_c, cls_scores, cls_boxes);

  // global top-100 per image + final output assembly
  topk_kernel<<<dim3(NUM_IMAGES), blk, 0, stream>>>(cls_scores, cls_boxes, out);
}
